// MultiHeadAttention_61924838473791
// MI455X (gfx1250) — matrix operand order repaired and checked by reading, not compiled, not run on hardware
//
#include <hip/hip_runtime.h>
#include <hip/hip_bf16.h>
#include <math.h>

// ---------------- problem constants ----------------
#define EMB   768
#define SEQ   2048
#define NHEAD 12
#define HDIM  64
#define NBAT  2
#define MTOT  (NBAT * SEQ)   // 4096
#define NQKV  (3 * EMB)      // 2304
#define QSCALE 0.125f        // 1/sqrt(64), exact power of two

typedef __attribute__((ext_vector_type(16))) __bf16 v16bf;
typedef __attribute__((ext_vector_type(8)))  __bf16 bf16x8;
typedef __attribute__((ext_vector_type(8)))  float  v8f;

// ---------------- WMMA wrapper (bf16 in, f32 accum) ----------------
__device__ __forceinline__ v8f wmma_bf16(v16bf a, v16bf b, v8f c) {
  // (neg_a, A, neg_b, B, c_mod, C, reuse_a, reuse_b)
  return __builtin_amdgcn_wmma_f32_16x16x32_bf16(false, a, false, b, (short)0, c,
                                                 false, false);
}

// ---------------- operand fragment loaders ----------------
// A operand: 16(M) x 32(K), src row-major [*, ld] bf16.
// ISA layout: lane L: M = L&15; K segments {s..s+7} and {s+16..s+23}, s=(L>>4)*8.
__device__ __forceinline__ v16bf load_a(const __bf16* __restrict__ src, int ld,
                                        int row0, int ko) {
  const int lane = threadIdx.x & 31;
  const __bf16* p = src + (size_t)(row0 + (lane & 15)) * ld + ko + ((lane >> 4) << 3);
  union { v16bf v; bf16x8 h[2]; } u;
  u.h[0] = *reinterpret_cast<const bf16x8*>(p);
  u.h[1] = *reinterpret_cast<const bf16x8*>(p + 16);
  return u.v;
}

// B operand: 32(K) x 16(N), supplied as N x K row-major (i.e. B^T in memory).
// ISA layout: lane L: N = L&15; K segments {s..s+7} and {s+16..s+23}, s=(L>>4)*8.
__device__ __forceinline__ v16bf load_b(const __bf16* __restrict__ src, int ld,
                                        int row0, int ko) {
  const int lane = threadIdx.x & 31;
  const __bf16* p = src + (size_t)(row0 + (lane & 15)) * ld + ko + ((lane >> 4) << 3);
  union { v16bf v; bf16x8 h[2]; } u;
  u.h[0] = *reinterpret_cast<const bf16x8*>(p);
  u.h[1] = *reinterpret_cast<const bf16x8*>(p + 16);
  return u.v;
}

// ---------------- fp32 -> bf16 conversion ----------------
__global__ void cvt_f32_bf16_kernel(const float* __restrict__ src,
                                    __bf16* __restrict__ dst, int n) {
  for (int i = blockIdx.x * blockDim.x + threadIdx.x; i < n;
       i += gridDim.x * blockDim.x)
    dst[i] = (__bf16)src[i];
}

// ---------------- QKV projection: qkv = x @ Wqkv^T + b, scatter to heads ----
// X  : [MTOT, EMB] bf16 row-major
// W  : [NQKV, EMB] bf16 row-major (N x K form -> direct B operand)
// Qh : [B,H,T,D] bf16 (pre-scaled by QSCALE)
// Kh : [B,H,T,D] bf16
// Vt : [B,H,D,T] bf16 (transposed V)
__global__ void qkv_gemm_kernel(const __bf16* __restrict__ X,
                                const __bf16* __restrict__ W,
                                const float*  __restrict__ bias,
                                __bf16* __restrict__ Qh,
                                __bf16* __restrict__ Kh,
                                __bf16* __restrict__ Vt) {
  const int wave = (blockIdx.x * blockDim.x + threadIdx.x) >> 5;
  const int ntn  = NQKV / 32;             // 72
  const int tm   = wave / ntn;
  const int tn   = wave - tm * ntn;
  if (tm >= MTOT / 32) return;
  const int m0 = tm * 32, n0 = tn * 32;
  const int lane = threadIdx.x & 31;

  v8f c00 = {}, c01 = {}, c10 = {}, c11 = {};
  for (int ko = 0; ko < EMB; ko += 32) {
    if (ko + 32 < EMB) {
      __builtin_prefetch(X + (size_t)(m0 + (lane & 15)) * EMB + ko + 32, 0, 0);
      __builtin_prefetch(W + (size_t)(n0 + (lane & 15)) * EMB + ko + 32, 0, 0);
    }
    v16bf a0 = load_a(X, EMB, m0,      ko);
    v16bf a1 = load_a(X, EMB, m0 + 16, ko);
    v16bf b0 = load_b(W, EMB, n0,      ko);
    v16bf b1 = load_b(W, EMB, n0 + 16, ko);
    c00 = wmma_bf16(a0, b0, c00);
    c01 = wmma_bf16(a0, b1, c01);
    c10 = wmma_bf16(a1, b0, c10);
    c11 = wmma_bf16(a1, b1, c11);
  }

  // scatter with bias; C layout: elem i -> row = i + (lane>>4)*8, col = lane&15
  const int colL = lane & 15;
  const int rowH = (lane >> 4) << 3;
  v8f* ctab[2][2] = {{&c00, &c01}, {&c10, &c11}};
#pragma unroll
  for (int ti = 0; ti < 2; ++ti) {
#pragma unroll
    for (int tj = 0; tj < 2; ++tj) {
      const int n   = n0 + tj * 16 + colL;
      const float bn = bias[n];
      const int sec = n / EMB;            // 0:Q 1:K 2:V
      const int f   = n - sec * EMB;
      const int h   = f >> 6;             // head
      const int d   = f & 63;
      v8f cc = *ctab[ti][tj];
#pragma unroll
      for (int i = 0; i < 8; ++i) {
        const int m = m0 + ti * 16 + rowH + i;
        const int b = m >> 11;            // m / SEQ
        const int t = m & (SEQ - 1);
        const float v = cc[i] + bn;
        const size_t bh = (size_t)(b * NHEAD + h);
        if (sec == 0)
          Qh[(bh * SEQ + t) * HDIM + d] = (__bf16)(v * QSCALE);
        else if (sec == 1)
          Kh[(bh * SEQ + t) * HDIM + d] = (__bf16)v;
        else
          Vt[(bh * HDIM + d) * SEQ + t] = (__bf16)v;
      }
    }
  }
}

// ---------------- flash attention (causal, online softmax) ----------------
// One wave per 16-query tile of one (b,h). Computes S^T = K * Q^T so each
// lane owns one query column; P^T register layout == A-operand layout of P*V.
__global__ void attn_kernel(const __bf16* __restrict__ Qh,
                            const __bf16* __restrict__ Kh,
                            const __bf16* __restrict__ Vt,
                            __bf16* __restrict__ AO) {
  const int wave = (blockIdx.x * blockDim.x + threadIdx.x) >> 5;
  const int qt = wave & (SEQ / 16 - 1);   // 0..127
  const int bh = wave >> 7;               // 0..23
  if (bh >= NBAT * NHEAD) return;
  const int lane = threadIdx.x & 31;

  const __bf16* Q = Qh + (size_t)bh * SEQ * HDIM;
  const __bf16* K = Kh + (size_t)bh * SEQ * HDIM;
  const __bf16* V = Vt + (size_t)bh * HDIM * SEQ;
  const int q0 = qt * 16;

  // Q as B operand (N=queries, K=d), loaded once: two 32-d chunks
  const v16bf qb0 = load_b(Q, HDIM, q0, 0);
  const v16bf qb1 = load_b(Q, HDIM, q0, 32);

  v8f oc0 = {}, oc1 = {}, oc2 = {}, oc3 = {};
  float m_run = -__builtin_inff();
  float l_run = 0.0f;
  const int myq  = q0 + (lane & 15);      // this lane's query (global)
  const int hb   = (lane >> 4) << 3;      // 0 or 8
  const int kt_last = q0 >> 5;

  for (int kc = 0; kc <= kt_last; ++kc) {
    const int k0 = kc * 32;
    // S^T tiles: rows=keys, cols=queries.  A = K rows, B = Q.
    v8f s0 = {}, s1 = {};
    s0 = wmma_bf16(load_a(K, HDIM, k0,      0),  qb0, s0);
    s0 = wmma_bf16(load_a(K, HDIM, k0,      32), qb1, s0);
    s1 = wmma_bf16(load_a(K, HDIM, k0 + 16, 0),  qb0, s1);
    s1 = wmma_bf16(load_a(K, HDIM, k0 + 16, 32), qb1, s1);

    if (kc == kt_last) {                  // causal mask on diagonal chunk
      const int krow = k0 + hb;
#pragma unroll
      for (int i = 0; i < 8; ++i) {
        if (krow + i      > myq) s0[i] = -__builtin_inff();
        if (krow + 16 + i > myq) s1[i] = -__builtin_inff();
      }
    }

    // per-query online softmax (lane-local + one cross-half combine)
    float mloc = s0[0];
#pragma unroll
    for (int i = 0; i < 8; ++i) {
      mloc = fmaxf(mloc, s0[i]);
      mloc = fmaxf(mloc, s1[i]);
    }
    mloc = fmaxf(mloc, __shfl_xor(mloc, 16, 32));
    const float m_new = fmaxf(m_run, mloc);

    float p0[8], p1[8], rs = 0.0f;
#pragma unroll
    for (int i = 0; i < 8; ++i) {
      p0[i] = __expf(s0[i] - m_new);
      p1[i] = __expf(s1[i] - m_new);
      rs += p0[i] + p1[i];
    }
    rs += __shfl_xor(rs, 16, 32);
    const float alpha = __expf(m_run - m_new);
    l_run = l_run * alpha + rs;
    m_run = m_new;

    // broadcast alpha to O-accumulator rows (queries hb..hb+7)
    float fsc[8];
#pragma unroll
    for (int i = 0; i < 8; ++i) fsc[i] = __shfl(alpha, hb + i, 32);
#pragma unroll
    for (int i = 0; i < 8; ++i) {
      oc0[i] *= fsc[i]; oc1[i] *= fsc[i];
      oc2[i] *= fsc[i]; oc3[i] *= fsc[i];
    }

    // P^T registers -> A operand of P*V, direct per-lane bf16 convert
    v16bf pa;
#pragma unroll
    for (int i = 0; i < 8; ++i) {
      pa[i]     = (__bf16)p0[i];
      pa[i + 8] = (__bf16)p1[i];
    }
    oc0 = wmma_bf16(pa, load_b(V, SEQ, 0,  k0), oc0);
    oc1 = wmma_bf16(pa, load_b(V, SEQ, 16, k0), oc1);
    oc2 = wmma_bf16(pa, load_b(V, SEQ, 32, k0), oc2);
    oc3 = wmma_bf16(pa, load_b(V, SEQ, 48, k0), oc3);
  }

  // normalize and store to concat-head layout AO[b, t, h*64 + d]
  const float rinv = 1.0f / l_run;
  float fno[8];
#pragma unroll
  for (int i = 0; i < 8; ++i) fno[i] = __shfl(rinv, hb + i, 32);

  const int b = bh / NHEAD;
  const int h = bh - b * NHEAD;
  size_t base = ((size_t)b * SEQ + q0 + hb) * EMB + h * HDIM + (lane & 15);
  v8f* otab[4] = {&oc0, &oc1, &oc2, &oc3};
#pragma unroll
  for (int dt = 0; dt < 4; ++dt) {
    v8f o = *otab[dt];
#pragma unroll
    for (int i = 0; i < 8; ++i)
      AO[base + (size_t)i * EMB + dt * 16] = (__bf16)(o[i] * fno[i]);
  }
}

// ---------------- final projection: out = AO @ Wf^T (fp32 out) -------------
__global__ void final_gemm_kernel(const __bf16* __restrict__ AO,
                                  const __bf16* __restrict__ Wf,
                                  float* __restrict__ out) {
  const int wave = (blockIdx.x * blockDim.x + threadIdx.x) >> 5;
  const int ntn  = EMB / 32;              // 24
  const int tm   = wave / ntn;
  const int tn   = wave - tm * ntn;
  if (tm >= MTOT / 32) return;
  const int m0 = tm * 32, n0 = tn * 32;
  const int lane = threadIdx.x & 31;

  v8f c00 = {}, c01 = {}, c10 = {}, c11 = {};
  for (int ko = 0; ko < EMB; ko += 32) {
    if (ko + 32 < EMB) {
      __builtin_prefetch(AO + (size_t)(m0 + (lane & 15)) * EMB + ko + 32, 0, 0);
      __builtin_prefetch(Wf + (size_t)(n0 + (lane & 15)) * EMB + ko + 32, 0, 0);
    }
    v16bf a0 = load_a(AO, EMB, m0,      ko);
    v16bf a1 = load_a(AO, EMB, m0 + 16, ko);
    v16bf b0 = load_b(Wf, EMB, n0,      ko);
    v16bf b1 = load_b(Wf, EMB, n0 + 16, ko);
    c00 = wmma_bf16(a0, b0, c00);
    c01 = wmma_bf16(a0, b1, c01);
    c10 = wmma_bf16(a1, b0, c10);
    c11 = wmma_bf16(a1, b1, c11);
  }

  const int colL = lane & 15;
  const int rowH = (lane >> 4) << 3;
  v8f* ctab[2][2] = {{&c00, &c01}, {&c10, &c11}};
#pragma unroll
  for (int ti = 0; ti < 2; ++ti)
#pragma unroll
    for (int tj = 0; tj < 2; ++tj) {
      v8f cc = *ctab[ti][tj];
#pragma unroll
      for (int i = 0; i < 8; ++i)
        out[(size_t)(m0 + ti * 16 + rowH + i) * EMB + n0 + tj * 16 + colL] = cc[i];
    }
}

// ---------------- host launcher ----------------
extern "C" void kernel_launch(void* const* d_in, const int* in_sizes, int n_in,
                              void* d_out, int out_size, void* d_ws, size_t ws_size,
                              hipStream_t stream) {
  const float* x       = (const float*)d_in[0];  // [2,2048,768]
  const float* w_qkv   = (const float*)d_in[1];  // [2304,768]
  const float* b_qkv   = (const float*)d_in[2];  // [2304]
  const float* w_final = (const float*)d_in[3];  // [768,768]
  float* out = (float*)d_out;

  char* ws = (char*)d_ws;
  size_t off = 0;
  __bf16* xb   = (__bf16*)(ws + off); off += (size_t)MTOT * EMB * 2;      // 6.0 MB
  __bf16* wqb  = (__bf16*)(ws + off); off += (size_t)NQKV * EMB * 2;      // 3.4 MB
  __bf16* wfb  = (__bf16*)(ws + off); off += (size_t)EMB * EMB * 2;       // 1.1 MB
  __bf16* Qh   = (__bf16*)(ws + off); off += (size_t)MTOT * EMB * 2;
  __bf16* Kh   = (__bf16*)(ws + off); off += (size_t)MTOT * EMB * 2;
  __bf16* Vt   = (__bf16*)(ws + off); off += (size_t)MTOT * EMB * 2;
  __bf16* AO   = (__bf16*)(ws + off); off += (size_t)MTOT * EMB * 2;      // ~36 MB total

  // 1) fp32 -> bf16 conversions
  cvt_f32_bf16_kernel<<<2048, 256, 0, stream>>>(x,       xb,  MTOT * EMB);
  cvt_f32_bf16_kernel<<<1024, 256, 0, stream>>>(w_qkv,   wqb, NQKV * EMB);
  cvt_f32_bf16_kernel<<<512,  256, 0, stream>>>(w_final, wfb, EMB * EMB);

  // 2) QKV projection + head scatter: (4096/32)*(2304/32)=9216 waves, 8/block
  qkv_gemm_kernel<<<9216 / 8, 256, 0, stream>>>(xb, wqb, b_qkv, Qh, Kh, Vt);

  // 3) causal flash attention: 24*(2048/16)=3072 waves, 8/block
  attn_kernel<<<3072 / 8, 256, 0, stream>>>(Qh, Kh, Vt, AO);

  // 4) output projection: (4096/32)*(768/32)=3072 waves
  final_gemm_kernel<<<3072 / 8, 256, 0, stream>>>(AO, wfb, out);
}
